// Model_13365938226066
// MI455X (gfx1250) — matrix operand order repaired and checked by reading, not compiled, not run on hardware
//
#include <hip/hip_runtime.h>
#include <stdint.h>

typedef __attribute__((ext_vector_type(16))) __bf16 v16bf;
typedef __attribute__((ext_vector_type(8)))  float  v8f;

#define B_    16
#define S_    2048
#define D_    128
#define KT    64        // keys per main-loop iteration
#define PSTR  72        // P LDS row stride (bf16): 144B rows, 16B aligned

__device__ __forceinline__ unsigned pack_bf2(float a, float b) {
  union { __bf16 h[2]; unsigned u; } r;
  r.h[0] = (__bf16)a; r.h[1] = (__bf16)b;
  return r.u;
}

__device__ __forceinline__ uint32_t hmix(uint32_t x) {   // deterministic dropout hash
  x ^= x >> 16; x *= 0x7feb352dU;
  x ^= x >> 15; x *= 0x846ca68bU;
  x ^= x >> 16; return x;
}

// =====================================================================
// Pre-pass: one-time f32 -> bf16 conversion into workspace.
//   Qbf[b][s][d]            = bf16(Q * log2e / scale)   (row-major)
//   Kbf[b][s][d]            = bf16(K)                   (row-major)
//   Vbf[b][kt][d][keypos]   = bf16(V) transposed + key-permuted tiles
//     keypos(key = t*16+i)  = (i>>3)*32 + (i&7)*4 + t
// =====================================================================
__global__ __launch_bounds__(256)
void preconv(const float* __restrict__ Qf, const float* __restrict__ Kf,
             const float* __restrict__ Vf, const float* __restrict__ scale,
             __bf16* __restrict__ Qb, __bf16* __restrict__ Kb, __bf16* __restrict__ Vb)
{
  const unsigned u = blockIdx.x * 256 + threadIdx.x;    // [0, 1M)
  const float qs = 1.44269504088896f / scale[0];
  {
    const float4 f = *(const float4*)(Qf + (size_t)u * 4);
    *(uint2*)(Qb + (size_t)u * 4) =
        make_uint2(pack_bf2(f.x * qs, f.y * qs), pack_bf2(f.z * qs, f.w * qs));
    const float4 g = *(const float4*)(Kf + (size_t)u * 4);
    *(uint2*)(Kb + (size_t)u * 4) =
        make_uint2(pack_bf2(g.x, g.y), pack_bf2(g.z, g.w));
  }
  {
    const int dim = u & 127;
    const unsigned r  = u >> 7;
    const int i   = r & 15;
    const int kt  = (r >> 4) & 31;
    const int b   = r >> 9;
    const float* vt = Vf + ((size_t)b * S_ + (size_t)kt * KT) * D_ + dim;
    const float f0 = vt[(0*16 + i) * D_];
    const float f1 = vt[(1*16 + i) * D_];
    const float f2 = vt[(2*16 + i) * D_];
    const float f3 = vt[(3*16 + i) * D_];
    const int keypos = ((i >> 3) << 5) + ((i & 7) << 2);
    *(uint2*)(Vb + (((size_t)(b * 32 + kt) * D_ + dim) << 6) + keypos) =
        make_uint2(pack_bf2(f0, f1), pack_bf2(f2, f3));
  }
}

// =====================================================================
// Main flash-attention kernel: operands pre-packed bf16 in WMMA layouts.
// QK interleaves both M-tiles over a shared ping-pong B fragment
// (1 B-load : 2 WMMAs) while keeping only one chunk of Q live.
// =====================================================================
__global__ __launch_bounds__(128)
void fattn_fwd2(const __bf16* __restrict__ Qb, const __bf16* __restrict__ Kb,
                const __bf16* __restrict__ Vb, float* __restrict__ O)
{
  __shared__ __bf16 Plds[4][32 * PSTR];     // per-wave P relayout regions

  const int tid  = threadIdx.x;
  const int wave = tid >> 5;
  const int lane = tid & 31;
  const int half = lane >> 4;
  const int n    = lane & 15;

  const int bb   = blockIdx.x >> 4;
  const int qblk = blockIdx.x & 15;
  const int q0   = qblk * 128 + wave * 32;

  v8f acc[2][8];
  v8f zero = {};
  #pragma unroll
  for (int m = 0; m < 2; ++m)
    #pragma unroll
    for (int t = 0; t < 8; ++t) acc[m][t] = zero;
  float mrow[2][8], lrow[2][8];
  #pragma unroll
  for (int m = 0; m < 2; ++m)
    #pragma unroll
    for (int v = 0; v < 8; ++v) { mrow[m][v] = -3.0e38f; lrow[m][v] = 0.0f; }

  const __bf16* qr0 = Qb + ((size_t)bb * S_ + (q0 + n)) * D_ + 16*half;      // m=0 row
  const __bf16* qr1 = qr0 + (size_t)16 * D_;                                  // m=1 row
  const __bf16* kbb = Kb + ((size_t)bb * S_) * D_;
  const __bf16* vbb = Vb + ((size_t)bb * 32) * D_ * 64;

  #pragma unroll 1
  for (int kt = 0; kt < S_/KT; ++kt) {
    const __bf16* kbase = kbb + (size_t)kt * KT * D_ + (size_t)n * D_ + 16*half;
    const __bf16* vbase = vbb + ((size_t)kt * D_) * 64 + (size_t)n * 64 + 16*half;
    __builtin_prefetch(kbase + KT * D_, 0, 0);
    __builtin_prefetch(vbase + (size_t)D_ * 64, 0, 0);

    // ---- S = Q K^T : both M-tiles share each B fragment ----
    v8f sC[2][4];
    #pragma unroll
    for (int m = 0; m < 2; ++m)
      #pragma unroll
      for (int t = 0; t < 4; ++t) sC[m][t] = zero;

    v16bf bk[2];
    {
      uint4* p = (uint4*)&bk[0];
      p[0] = *(const uint4*)kbase;
      p[1] = *(const uint4*)(kbase + 8);
    }
    #pragma unroll
    for (int c = 0; c < 4; ++c) {
      v16bf aq0, aq1;                       // Q A-fragments for this chunk only
      {
        uint4* p = (uint4*)&aq0;
        p[0] = *(const uint4*)(qr0 + c*32);
        p[1] = *(const uint4*)(qr0 + c*32 + 8);
        uint4* q = (uint4*)&aq1;
        q[0] = *(const uint4*)(qr1 + c*32);
        q[1] = *(const uint4*)(qr1 + c*32 + 8);
      }
      #pragma unroll
      for (int t = 0; t < 4; ++t) {
        const int idx = c*4 + t;
        if (idx < 15) {
          const int c2 = (idx + 1) >> 2, t2 = (idx + 1) & 3;
          uint4* p = (uint4*)&bk[(idx + 1) & 1];
          const __bf16* kr = kbase + (size_t)(t2 * 16) * D_ + c2*32;
          p[0] = *(const uint4*)kr;  p[1] = *(const uint4*)(kr + 8);
        }
        sC[0][t] = __builtin_amdgcn_wmma_f32_16x16x32_bf16(false, aq0, false,
                              bk[idx & 1], (short)0, sC[0][t], false, false);
        sC[1][t] = __builtin_amdgcn_wmma_f32_16x16x32_bf16(false, aq1, false,
                              bk[idx & 1], (short)0, sC[1][t], false, false);
      }
    }

    // ---- online softmax (exp2 domain) + dropout, P -> LDS ----
    #pragma unroll
    for (int m = 0; m < 2; ++m) {
      float corr[8];
      #pragma unroll
      for (int v = 0; v < 8; ++v) {
        float mv = fmaxf(fmaxf(sC[m][0][v], sC[m][1][v]),
                         fmaxf(sC[m][2][v], sC[m][3][v]));
        #pragma unroll
        for (int off = 8; off >= 1; off >>= 1)
          mv = fmaxf(mv, __shfl_xor(mv, off, 32));
        const float mnew = fmaxf(mrow[m][v], mv);
        corr[v] = __builtin_amdgcn_exp2f(mrow[m][v] - mnew);
        mrow[m][v] = mnew;
        float p[4], ps = 0.0f;
        #pragma unroll
        for (int t = 0; t < 4; ++t) {
          p[t] = __builtin_amdgcn_exp2f(sC[m][t][v] - mnew);
          ps += p[t];
        }
        #pragma unroll
        for (int off = 8; off >= 1; off >>= 1)
          ps += __shfl_xor(ps, off, 32);
        lrow[m][v] = lrow[m][v] * corr[v] + ps;
        // dropout: 1 hash -> 4 byte decisions (keep if byte < 51 ~= p=0.2), 5x
        const int row = m*16 + v + 8*half;
        const unsigned h = hmix((unsigned)(bb * S_ + q0 + row) * (unsigned)S_ +
                                (unsigned)(kt * KT + n));
        float d[4];
        #pragma unroll
        for (int t = 0; t < 4; ++t)
          d[t] = (((h >> (8*t)) & 0xFFu) < 51u) ? p[t] * 5.0f : 0.0f;
        const int keypos = ((n >> 3) << 5) + ((n & 7) << 2);
        *(uint2*)&Plds[wave][row * PSTR + keypos] =
            make_uint2(pack_bf2(d[0], d[1]), pack_bf2(d[2], d[3]));
      }
      #pragma unroll
      for (int t = 0; t < 8; ++t)
        #pragma unroll
        for (int v = 0; v < 8; ++v)
          acc[m][t][v] *= corr[v];
    }

    // first V fragment: global load issued before the LDS fence
    v16bf bv[2];
    {
      uint4* p = (uint4*)&bv[0];
      p[0] = *(const uint4*)vbase;
      p[1] = *(const uint4*)(vbase + 8);
    }
    asm volatile("s_wait_dscnt 0" ::: "memory");   // P stores -> P loads

    // ---- P A-fragments from LDS ----
    v16bf pf[2][2];
    #pragma unroll
    for (int m = 0; m < 2; ++m)
      #pragma unroll
      for (int f = 0; f < 2; ++f) {
        uint4* p = (uint4*)&pf[m][f];
        const int base = (m*16 + n) * PSTR + f*32 + 16*half;
        p[0] = *(const uint4*)&Plds[wave][base];
        p[1] = *(const uint4*)&Plds[wave][base + 8];
      }

    // ---- acc += P @ V : each V fragment feeds both M-tiles ----
    #pragma unroll
    for (int idx = 0; idx < 16; ++idx) {
      const int t = idx >> 1, f = idx & 1;
      if (idx < 15) {
        const int t2 = (idx + 1) >> 1, f2 = (idx + 1) & 1;
        uint4* p = (uint4*)&bv[(idx + 1) & 1];
        const __bf16* vr = vbase + (size_t)(t2 * 16) * 64 + f2*32;
        p[0] = *(const uint4*)vr;  p[1] = *(const uint4*)(vr + 8);
      }
      #pragma unroll
      for (int m = 0; m < 2; ++m)
        acc[m][t] = __builtin_amdgcn_wmma_f32_16x16x32_bf16(false, pf[m][f], false,
                                  bv[idx & 1], (short)0, acc[m][t], false, false);
    }
  }

  // ---- normalize and store fp32 ----
  float* orow = O + (size_t)bb * S_ * D_;
  #pragma unroll
  for (int m = 0; m < 2; ++m) {
    float rl[8];
    #pragma unroll
    for (int v = 0; v < 8; ++v) rl[v] = __builtin_amdgcn_rcpf(lrow[m][v]);
    #pragma unroll
    for (int t = 0; t < 8; ++t)
      #pragma unroll
      for (int v = 0; v < 8; ++v)
        orow[(size_t)(q0 + m*16 + v + 8*half) * D_ + t*16 + n] = acc[m][t][v] * rl[v];
  }
}

// =====================================================================
// Fallback (self-contained, converts in-kernel) when d_ws is too small.
// =====================================================================
#define QSTR  136
#define KSTR  136
#define VSTR  72

__global__ __launch_bounds__(128)
void fattn_fwd_fb(const float* __restrict__ Q, const float* __restrict__ K,
                  const float* __restrict__ V, const float* __restrict__ scale,
                  float* __restrict__ O)
{
  __shared__ __bf16 Qlds[4][32 * QSTR];
  __shared__ __bf16 Klds[KT * KSTR];
  __shared__ __bf16 Vlds[D_ * VSTR];
  __shared__ __bf16 Plds[4][32 * PSTR];

  const int tid = threadIdx.x, wave = tid >> 5, lane = tid & 31;
  const int half = lane >> 4, n = lane & 15;
  const int bb = blockIdx.x >> 4, qblk = blockIdx.x & 15;
  const int q0 = qblk * 128 + wave * 32;
  const float qs = 1.44269504088896f / scale[0];
  const size_t boff = (size_t)bb * S_ * D_;

  {
    __bf16* qw = Qlds[wave];
    #pragma unroll
    for (int i = 0; i < 32; ++i) {
      const int u = i * 32 + lane;
      const int row = u >> 5, dim4 = (u & 31) << 2;
      const float4 f = *(const float4*)(Q + boff + (size_t)(q0 + row) * D_ + dim4);
      *(uint2*)&qw[row * QSTR + dim4] =
          make_uint2(pack_bf2(f.x * qs, f.y * qs), pack_bf2(f.z * qs, f.w * qs));
    }
  }
  v8f acc[2][8]; v8f zero = {};
  #pragma unroll
  for (int m = 0; m < 2; ++m)
    #pragma unroll
    for (int t = 0; t < 8; ++t) acc[m][t] = zero;
  float mrow[2][8], lrow[2][8];
  #pragma unroll
  for (int m = 0; m < 2; ++m)
    #pragma unroll
    for (int v = 0; v < 8; ++v) { mrow[m][v] = -3.0e38f; lrow[m][v] = 0.0f; }

  #pragma unroll 1
  for (int kt = 0; kt < S_/KT; ++kt) {
    __syncthreads();
    {
      const float* kbase = K + boff + (size_t)kt * KT * D_;
      #pragma unroll
      for (int i = 0; i < 16; ++i) {
        const int lin = i * 512 + tid * 4;
        const int key = lin >> 7, dim = lin & 127;
        const float4 kv = *(const float4*)(kbase + lin);
        *(uint2*)&Klds[key * KSTR + dim] =
            make_uint2(pack_bf2(kv.x, kv.y), pack_bf2(kv.z, kv.w));
      }
      const float* vbase = V + boff + (size_t)kt * KT * D_;
      #pragma unroll
      for (int i = 0; i < 16; ++i) {
        const float f0 = vbase[(0*16 + i) * D_ + tid];
        const float f1 = vbase[(1*16 + i) * D_ + tid];
        const float f2 = vbase[(2*16 + i) * D_ + tid];
        const float f3 = vbase[(3*16 + i) * D_ + tid];
        const int keypos = ((i >> 3) << 5) + ((i & 7) << 2);
        *(uint2*)&Vlds[tid * VSTR + keypos] =
            make_uint2(pack_bf2(f0, f1), pack_bf2(f2, f3));
      }
    }
    __syncthreads();
    #pragma unroll
    for (int m = 0; m < 2; ++m) {
      v8f sC[4];
      #pragma unroll
      for (int t = 0; t < 4; ++t) sC[t] = zero;
      #pragma unroll
      for (int c = 0; c < 4; ++c) {
        v16bf aq;
        { uint4* p = (uint4*)&aq;
          const int base = (m*16 + n) * QSTR + c*32 + 16*half;
          p[0] = *(const uint4*)&Qlds[wave][base];
          p[1] = *(const uint4*)&Qlds[wave][base + 8]; }
        #pragma unroll
        for (int t = 0; t < 4; ++t) {
          v16bf bk;
          { uint4* p = (uint4*)&bk;
            const int base = (t*16 + n) * KSTR + c*32 + 16*half;
            p[0] = *(const uint4*)&Klds[base];
            p[1] = *(const uint4*)&Klds[base + 8]; }
          sC[t] = __builtin_amdgcn_wmma_f32_16x16x32_bf16(false, aq, false, bk,
                                                          (short)0, sC[t], false, false);
        }
      }
      float corr[8];
      #pragma unroll
      for (int v = 0; v < 8; ++v) {
        float mv = fmaxf(fmaxf(sC[0][v], sC[1][v]), fmaxf(sC[2][v], sC[3][v]));
        #pragma unroll
        for (int off = 8; off >= 1; off >>= 1) mv = fmaxf(mv, __shfl_xor(mv, off, 32));
        const float mnew = fmaxf(mrow[m][v], mv);
        corr[v] = __builtin_amdgcn_exp2f(mrow[m][v] - mnew);
        mrow[m][v] = mnew;
        float p[4], ps = 0.0f;
        #pragma unroll
        for (int t = 0; t < 4; ++t) { p[t] = __builtin_amdgcn_exp2f(sC[t][v] - mnew); ps += p[t]; }
        #pragma unroll
        for (int off = 8; off >= 1; off >>= 1) ps += __shfl_xor(ps, off, 32);
        lrow[m][v] = lrow[m][v] * corr[v] + ps;
        const int row = m*16 + v + 8*half;
        const unsigned h = hmix((unsigned)(bb * S_ + q0 + row) * (unsigned)S_ +
                                (unsigned)(kt * KT + n));
        float d[4];
        #pragma unroll
        for (int t = 0; t < 4; ++t)
          d[t] = (((h >> (8*t)) & 0xFFu) < 51u) ? p[t] * 5.0f : 0.0f;
        const int keypos = ((n >> 3) << 5) + ((n & 7) << 2);
        *(uint2*)&Plds[wave][row * PSTR + keypos] =
            make_uint2(pack_bf2(d[0], d[1]), pack_bf2(d[2], d[3]));
      }
      #pragma unroll
      for (int t = 0; t < 8; ++t)
        #pragma unroll
        for (int v = 0; v < 8; ++v) acc[m][t][v] *= corr[v];
    }
    asm volatile("s_wait_dscnt 0" ::: "memory");
    v16bf pf[2][2];
    #pragma unroll
    for (int m = 0; m < 2; ++m)
      #pragma unroll
      for (int f = 0; f < 2; ++f) {
        uint4* p = (uint4*)&pf[m][f];
        const int base = (m*16 + n) * PSTR + f*32 + 16*half;
        p[0] = *(const uint4*)&Plds[wave][base];
        p[1] = *(const uint4*)&Plds[wave][base + 8];
      }
    #pragma unroll
    for (int t = 0; t < 8; ++t)
      #pragma unroll
      for (int f = 0; f < 2; ++f) {
        v16bf bv;
        { uint4* p = (uint4*)&bv;
          const int base = (t*16 + n) * VSTR + f*32 + 16*half;
          p[0] = *(const uint4*)&Vlds[base];
          p[1] = *(const uint4*)&Vlds[base + 8]; }
        #pragma unroll
        for (int m = 0; m < 2; ++m)
          acc[m][t] = __builtin_amdgcn_wmma_f32_16x16x32_bf16(false, pf[m][f], false,
                                    bv, (short)0, acc[m][t], false, false);
      }
  }
  float* orow = O + boff;
  #pragma unroll
  for (int m = 0; m < 2; ++m) {
    float rl[8];
    #pragma unroll
    for (int v = 0; v < 8; ++v) rl[v] = __builtin_amdgcn_rcpf(lrow[m][v]);
    #pragma unroll
    for (int t = 0; t < 8; ++t)
      #pragma unroll
      for (int v = 0; v < 8; ++v)
        orow[(size_t)(q0 + m*16 + v + 8*half) * D_ + t*16 + n] = acc[m][t][v] * rl[v];
  }
}

extern "C" void kernel_launch(void* const* d_in, const int* in_sizes, int n_in,
                              void* d_out, int out_size, void* d_ws, size_t ws_size,
                              hipStream_t stream) {
  (void)in_sizes; (void)n_in; (void)out_size;
  const float* q  = (const float*)d_in[0];
  const float* k  = (const float*)d_in[1];
  const float* v  = (const float*)d_in[2];
  const float* sc = (const float*)d_in[3];
  float* out = (float*)d_out;
  const size_t N = (size_t)B_ * S_ * D_;          // 4,194,304 elements
  if (ws_size >= 3 * N * sizeof(__bf16)) {
    __bf16* qb = (__bf16*)d_ws;
    __bf16* kb = qb + N;
    __bf16* vb = kb + N;
    preconv<<<dim3((unsigned)(N / 4 / 256)), dim3(256), 0, stream>>>(q, k, v, sc, qb, kb, vb);
    fattn_fwd2<<<dim3(B_ * (S_ / 128)), dim3(128), 0, stream>>>(qb, kb, vb, out);
  } else {
    fattn_fwd_fb<<<dim3(B_ * (S_ / 128)), dim3(128), 0, stream>>>(q, k, v, sc, out);
  }
}
